// LocalAttention_18786186953466
// MI455X (gfx1250) — compile-verified, matrix operand order checked by reading, not run on hardware
//
#include <hip/hip_runtime.h>
#include <hip/hip_bf16.h>

typedef unsigned short u16;
typedef __attribute__((ext_vector_type(16))) __bf16 v16bf;
typedef __attribute__((ext_vector_type(8)))  float  v8f;

#define HEADS 12
#define HD    32          // head dim
#define NW    49          // tokens per 7x7 window
#define NP    64          // padded tokens
#define IMG   112
#define L     (IMG*IMG)   // 12544
#define DIM   384
#define NB    16          // batch
#define NF4   (NW * HD / 4)   // 392 float4 chunks per tensor per (window, head)

struct __align__(16) U128 { unsigned int x, y, z, w; };
struct __align__(16) F4   { float x, y, z, w; };
struct __align__(8)  BF4  { u16 a, b, c, d; };

union Frag {
    v16bf v;
    u16   u[16];
    U128  q[2];
};

// round-to-nearest-even f32 -> bf16 bits
__device__ __forceinline__ u16 f2bf(float f) {
    unsigned int u = __float_as_uint(f);
    u += 0x7FFFu + ((u >> 16) & 1u);
    return (u16)(u >> 16);
}

// Build one 16-bit A/B WMMA fragment from a row-major LDS row.
// Per CDNA5 ISA (16-bit A 16x32 layout): lanes 0-15 hold K {0..7,16..23},
// lanes 16-31 hold K {8..15,24..31}; frag elems e<8 -> K=half*8+e,
// e>=8 -> K=16+half*8+(e-8).  Two contiguous 16B loads per lane.
__device__ __forceinline__ v16bf frag_row(const u16* rowPtr, int half) {
    Frag f;
    f.q[0] = *(const U128*)(rowPtr + half * 8);
    f.q[1] = *(const U128*)(rowPtr + 16 + half * 8);
    return f.v;
}

__global__ __launch_bounds__(32)
void lwin_attn_lepe_kernel(const float* __restrict__ qkv,
                           const float* __restrict__ conv_w,
                           const float* __restrict__ conv_b,
                           float* __restrict__ out)
{
    __shared__ __align__(16) u16   sq[NP * HD];    // bf16 q (scaled), zero padded
    __shared__ __align__(16) u16   sk[NP * HD];    // bf16 k, zero padded
    __shared__ __align__(16) u16   sv[NP * HD];    // bf16 v, zero padded
    __shared__ __align__(16) float svf[NW * HD];   // f32 v for LePE conv
    __shared__ __align__(16) float sS[NP * NP];    // f32 scores; later reused for X
    __shared__ __align__(16) u16   sP[NP * NP];    // bf16 probabilities

    const int lane = threadIdx.x;       // 0..31
    const int lr   = lane & 15;
    const int half = lane >> 4;

    const int bid  = blockIdx.x;
    const int head = bid % HEADS;
    const int win  = bid / HEADS;       // 0..4095
    const int b    = win >> 8;          // / 256 windows per image
    const int wrem = win & 255;
    const int bh   = wrem >> 4;         // window row  (16 per image)
    const int bw   = wrem & 15;         // window col

    const size_t imgBase    = (size_t)b * L;        // token base of image
    const size_t tensStride = (size_t)NB * L * DIM; // one qkv tensor

    const float scale = 0.17677669529663687f;       // 32^-0.5

    // ------------------------------------------------------------------
    // Load q,k,v with float4 (global_load_b128): lane f covers token
    // t = f/8, channels 4*(f%8)..+3.  Fully coalesced 512B/instruction.
    // ------------------------------------------------------------------
    for (int it = 0; it < (NF4 + 31) / 32; ++it) {
        int f = it * 32 + lane;
        if (f < NF4) {
            int t   = f >> 3;            // token 0..48
            int ch4 = (f & 7) * 4;       // channel base within head
            int lh = t / 7, lw = t - lh * 7;
            size_t pix = (size_t)(bh * 7 + lh) * IMG + (bw * 7 + lw);
            size_t idx = (imgBase + pix) * DIM + head * HD + ch4;
            F4 qf = *(const F4*)(qkv + idx);
            F4 kf = *(const F4*)(qkv + tensStride + idx);
            F4 vf = *(const F4*)(qkv + 2 * tensStride + idx);
            BF4 qb = { f2bf(qf.x * scale), f2bf(qf.y * scale),
                       f2bf(qf.z * scale), f2bf(qf.w * scale) };
            BF4 kb4 = { f2bf(kf.x), f2bf(kf.y), f2bf(kf.z), f2bf(kf.w) };
            BF4 vb4 = { f2bf(vf.x), f2bf(vf.y), f2bf(vf.z), f2bf(vf.w) };
            *(BF4*)(sq + t * HD + ch4)  = qb;
            *(BF4*)(sk + t * HD + ch4)  = kb4;
            *(BF4*)(sv + t * HD + ch4)  = vb4;
            *(F4*)(svf + t * HD + ch4)  = vf;
        }
    }
    for (int t = NW; t < NP; ++t) {      // zero pad rows 49..63
        sq[t * HD + lane] = 0;
        sk[t * HD + lane] = 0;
        sv[t * HD + lane] = 0;
    }
    __syncthreads();

    // ------------------------------------------------------------------
    // S = (q*scale) @ k^T   -- 4x4 tiles of v_wmma_f32_16x16x32_bf16
    // A = q rows (M=query, K=dim); B = k^T (K=dim, N=key token) ->
    // b[e] = k[N][K], same row-major fetch as A.
    // ------------------------------------------------------------------
    v16bf kb[4];
#pragma unroll
    for (int nt = 0; nt < 4; ++nt)
        kb[nt] = frag_row(&sk[(nt * 16 + lr) * HD], half);

#pragma unroll
    for (int mt = 0; mt < 4; ++mt) {
        v16bf qa = frag_row(&sq[(mt * 16 + lr) * HD], half);
#pragma unroll
        for (int nt = 0; nt < 4; ++nt) {
            v8f acc = {0.f, 0.f, 0.f, 0.f, 0.f, 0.f, 0.f, 0.f};
            acc = __builtin_amdgcn_wmma_f32_16x16x32_bf16(
                false, qa, false, kb[nt], (short)0, acc, false, false);
            // C/D layout: VGPR j holds M = half*8 + j, N = lr
#pragma unroll
            for (int j = 0; j < 8; ++j)
                sS[(mt * 16 + half * 8 + j) * NP + nt * 16 + lr] = acc[j];
        }
    }
    __syncthreads();

    // ------------------------------------------------------------------
    // Row softmax (fp32), masked to 49 valid cols; P in bf16, zero pad.
    // Each lane owns rows {lane, lane+32}.
    // ------------------------------------------------------------------
#pragma unroll
    for (int rr = 0; rr < 2; ++rr) {
        int r = lane + rr * 32;
        if (r < NW) {
            float m = -3.4e38f;
            for (int j = 0; j < NW; ++j) m = fmaxf(m, sS[r * NP + j]);
            float s = 0.f;
            for (int j = 0; j < NW; ++j) s += __expf(sS[r * NP + j] - m);
            float inv = 1.f / s;
            for (int j = 0; j < NW; ++j)
                sP[r * NP + j] = f2bf(__expf(sS[r * NP + j] - m) * inv);
            for (int j = NW; j < NP; ++j) sP[r * NP + j] = 0;
        } else {
            for (int j = 0; j < NP; ++j) sP[r * NP + j] = 0;
        }
    }
    __syncthreads();

    // ------------------------------------------------------------------
    // X = P @ v  -- 4(M) x 2(N) tiles, K = 64 (2 k-steps).
    // B = v (K=token, N=channel): per-element gather from row-major sv.
    // ------------------------------------------------------------------
    float* sx = sS;     // reuse score buffer for X [NP][HD]

    v16bf vb[2][2];
#pragma unroll
    for (int kt = 0; kt < 2; ++kt)
#pragma unroll
        for (int nt = 0; nt < 2; ++nt) {
            Frag f;
#pragma unroll
            for (int e = 0; e < 16; ++e) {
                int K = ((e & 8) << 1) + half * 8 + (e & 7); // e<8 ? K : 16+K
                f.u[e] = sv[(kt * 32 + K) * HD + nt * 16 + lr];
            }
            vb[kt][nt] = f.v;
        }

#pragma unroll
    for (int mt = 0; mt < 4; ++mt) {
        v16bf pa0 = frag_row(&sP[(mt * 16 + lr) * NP], half);       // K 0..31
        v16bf pa1 = frag_row(&sP[(mt * 16 + lr) * NP + 32], half);  // K 32..63
#pragma unroll
        for (int nt = 0; nt < 2; ++nt) {
            v8f acc = {0.f, 0.f, 0.f, 0.f, 0.f, 0.f, 0.f, 0.f};
            acc = __builtin_amdgcn_wmma_f32_16x16x32_bf16(
                false, pa0, false, vb[0][nt], (short)0, acc, false, false);
            acc = __builtin_amdgcn_wmma_f32_16x16x32_bf16(
                false, pa1, false, vb[1][nt], (short)0, acc, false, false);
#pragma unroll
            for (int j = 0; j < 8; ++j)
                sx[(mt * 16 + half * 8 + j) * HD + nt * 16 + lr] = acc[j];
        }
    }
    __syncthreads();

    // ------------------------------------------------------------------
    // LePE: per-window 3x3 depthwise conv on fp32 v (lane = channel),
    // fused with the coalesced output store:  out = X + lepe.
    // ------------------------------------------------------------------
    const int c = head * HD + lane;     // lane owns one channel here
    float wreg[9];
#pragma unroll
    for (int i = 0; i < 9; ++i) wreg[i] = conv_w[(size_t)c * 9 + i];
    const float bias = conv_b[c];

    for (int t = 0; t < NW; ++t) {
        int lh = t / 7, lw = t - lh * 7;
        float acc = bias;
#pragma unroll
        for (int ky = 0; ky < 3; ++ky) {
            int y = lh + ky - 1;
            if (y < 0 || y >= 7) continue;
#pragma unroll
            for (int kx = 0; kx < 3; ++kx) {
                int x = lw + kx - 1;
                if (x < 0 || x >= 7) continue;
                acc += svf[(y * 7 + x) * HD + lane] * wreg[ky * 3 + kx];
            }
        }
        size_t pix = (size_t)(bh * 7 + lh) * IMG + (bw * 7 + lw);
        out[(imgBase + pix) * DIM + c] = sx[t * HD + lane] + acc;
    }
}

extern "C" void kernel_launch(void* const* d_in, const int* in_sizes, int n_in,
                              void* d_out, int out_size, void* d_ws, size_t ws_size,
                              hipStream_t stream) {
    const float* qkv    = (const float*)d_in[0];   // [3,16,12544,384] f32
    const float* conv_w = (const float*)d_in[1];   // [384,1,3,3] f32
    const float* conv_b = (const float*)d_in[2];   // [384] f32
    float* out = (float*)d_out;                    // [16,12544,384] f32

    const int nBlocks = 4096 * HEADS;              // one wave32 per (window, head)
    lwin_attn_lepe_kernel<<<dim3(nBlocks), dim3(32), 0, stream>>>(qkv, conv_w, conv_b, out);
}
